// DomainAwareLinear_6116033429683
// MI455X (gfx1250) — matrix operand order repaired and checked by reading, not compiled, not run on hardware
//
#include <hip/hip_runtime.h>
#include <hip/hip_bf16.h>

// Problem constants (match reference)
#define I_SIZE 2048
#define O_SIZE 2048
#define T_SIZE 2048
#define B_SIZE 16

// Tile config
#define BM 128          // rows of X (T dim) per block
#define BN 128          // cols of W (O dim) per block
#define BK 32           // K step (== bf16 WMMA K)
#define LDA 36          // LDS row stride (floats): 32 + 4 pad  (TDM: pad_interval=4, pad_amount=3)
#define LDB 132         // LDS row stride (floats): 128 + 4 pad (TDM: pad_interval=6, pad_amount=3)
#define KTILES (I_SIZE / BK)

typedef __attribute__((ext_vector_type(16))) __bf16       v16bf;
typedef __attribute__((ext_vector_type(8)))  float        v8f;
typedef __attribute__((ext_vector_type(4)))  int          v4i;
typedef __attribute__((ext_vector_type(4)))  unsigned int u32x4;
typedef __attribute__((ext_vector_type(8)))  int          i32x8;
typedef __attribute__((ext_vector_type(4)))  int          i32x4;

#define GLOBAL_AS __attribute__((address_space(1)))
#define LDS_AS    __attribute__((address_space(3)))
typedef GLOBAL_AS v4i* gv4i_t;
typedef LDS_AS    v4i* lv4i_t;

// ---- feature gates (device pass only) ----
#if defined(__AMDGCN__) && __has_builtin(__builtin_amdgcn_tensor_load_to_lds) && \
    __has_builtin(__builtin_amdgcn_s_wait_tensorcnt)
#define HAVE_TDM 1
#else
#define HAVE_TDM 0
#endif

#if defined(__AMDGCN__) && __has_builtin(__builtin_amdgcn_global_load_async_to_lds_b128)
#define HAVE_ASYNC_LDS 1
#else
#define HAVE_ASYNC_LDS 0
#endif

#if HAVE_TDM
// ROCm 7.2 clang-22: 5-arg builtin. amdgpu-toolchain (therock headers present): 6-arg.
#if __has_include(<hip/amd_detail/amd_gfx1250_TDM.h>)
#define TDM_LOAD(g0, g1, g2, g3) \
    __builtin_amdgcn_tensor_load_to_lds(g0, g1, g2, g3, (i32x8){0,0,0,0,0,0,0,0}, 0)
#else
#define TDM_LOAD(g0, g1, g2, g3) \
    __builtin_amdgcn_tensor_load_to_lds(g0, g1, g2, g3, 0)
#endif
#endif

__device__ __forceinline__ void wait_async_zero() {
#if defined(__AMDGCN__)
#if __has_builtin(__builtin_amdgcn_s_wait_asynccnt)
    __builtin_amdgcn_s_wait_asynccnt(0);
#else
    asm volatile("s_wait_asynccnt 0x0" ::: "memory");
#endif
#endif
}

__global__ __launch_bounds__(256)
void domain_linear_bf16x2_wmma(const float* __restrict__ x,
                               const int*   __restrict__ domain_id,
                               const float* __restrict__ fc_weight,
                               const float* __restrict__ bias_weight,
                               float*       __restrict__ out)
{
    extern __shared__ float smem[];
    float* ldsA = smem;                          // [2][BM * LDA]
    float* ldsB = smem + 2 * BM * LDA;           // [2][BK * LDB]

    const int tid   = threadIdx.x;
    const int lane  = tid & 31;
    const int wave  = tid >> 5;
    const int wy    = wave >> 2;                 // 0..1 : M wave coord (64 rows each)
    const int wx    = wave & 3;                  // 0..3 : N wave coord (32 cols each)
    const int hi16  = (lane >> 4) & 1;           // lane >= 16
    const int ln    = lane & 15;

    const int b     = blockIdx.z;
    const int tileM = blockIdx.y * BM;
    const int tileN = blockIdx.x * BN;
    const int dom   = domain_id[b];

    const float* xb = x           + (size_t)b   * T_SIZE * I_SIZE;
    const float* wb = fc_weight   + (size_t)dom * I_SIZE * O_SIZE;
    const float* bb = bias_weight + (size_t)dom * O_SIZE;
    float*       ob = out         + (size_t)b   * T_SIZE * O_SIZE;

    // ---- accumulators: 4 (M subtiles) x 2 (N subtiles), init with bias ----
    v8f acc[4][2];
#pragma unroll
    for (int ni = 0; ni < 2; ++ni) {
        const float bv = bb[tileN + wx * 32 + ni * 16 + ln];
#pragma unroll
        for (int mi = 0; mi < 4; ++mi)
#pragma unroll
            for (int r = 0; r < 8; ++r) acc[mi][ni][r] = bv;
    }

#if HAVE_TDM
    // ---- TDM staging: wave 0 issues one tensor DMA per tile per buffer ----
    // D# per ISA ch.8: group0 = {count, lds_addr, global_addr, type}, group1 holds
    // data_size / pad config / tensor dims / tile dims / strides. Groups 2-3 = 0 (2D tile).
    auto issue_tiles = [&](int buf, int k0) {
        if (wave == 0) {
            const i32x4 gz = {0, 0, 0, 0};
            {   // A tile: 32 (contig K) x 128 rows of x, pad 4 DWORDs per 32 -> LDA=36
                const unsigned long long ga =
                    (unsigned long long)(uintptr_t)(xb + (size_t)tileM * I_SIZE + k0);
                const unsigned la =
                    (unsigned)(uintptr_t)(LDS_AS void*)(ldsA + buf * (BM * LDA));
                const u32x4 g0 = { 1u, la, (unsigned)ga,
                                   (unsigned)((ga >> 32) & 0x01FFFFFFull) | 0x80000000u };
                const i32x8 g1 = {
                    (int)((2u << 16) | (1u << 20) | (4u << 22) | (3u << 25)), // 4B elems, pad on, 32dw/4dw
                    (int)((unsigned)(I_SIZE & 0xFFFF) << 16),                  // tensor_dim0 lo16 (2048)
                    (int)(((unsigned)I_SIZE >> 16) | ((unsigned)(T_SIZE & 0xFFFF) << 16)), // dim0 hi | dim1 lo
                    (int)(((unsigned)T_SIZE >> 16) | (32u << 16)),             // dim1 hi | tile_dim0=32
                    (int)128,                                                  // tile_dim1=128, tile_dim2=0
                    (int)I_SIZE,                                               // tensor_dim0_stride lo32
                    0, 0 };
                TDM_LOAD(g0, g1, gz, gz);
            }
            {   // B tile: 128 (contig N) x 32 rows of W, pad 4 DWORDs per 128 -> LDB=132
                const unsigned long long gb =
                    (unsigned long long)(uintptr_t)(wb + (size_t)k0 * O_SIZE + tileN);
                const unsigned lb =
                    (unsigned)(uintptr_t)(LDS_AS void*)(ldsB + buf * (BK * LDB));
                const u32x4 g0 = { 1u, lb, (unsigned)gb,
                                   (unsigned)((gb >> 32) & 0x01FFFFFFull) | 0x80000000u };
                const i32x8 g1 = {
                    (int)((2u << 16) | (1u << 20) | (6u << 22) | (3u << 25)), // 4B elems, pad on, 128dw/4dw
                    (int)((unsigned)(O_SIZE & 0xFFFF) << 16),                  // tensor_dim0 lo16 (2048)
                    (int)(((unsigned)O_SIZE >> 16) | ((unsigned)(I_SIZE & 0xFFFF) << 16)), // dim0 hi | dim1 lo
                    (int)(((unsigned)I_SIZE >> 16) | (128u << 16)),            // dim1 hi | tile_dim0=128
                    (int)32,                                                   // tile_dim1=32, tile_dim2=0
                    (int)O_SIZE,                                               // tensor_dim0_stride lo32
                    0, 0 };
                TDM_LOAD(g0, g1, gz, gz);
            }
        }
    };
    auto sync_tiles = [&]() {
        __builtin_amdgcn_s_wait_tensorcnt(0);   // no-op for non-issuing waves (count already 0)
        __syncthreads();
    };
#elif HAVE_ASYNC_LDS
    // ---- per-lane async staging ----
    auto issue_tiles = [&](int buf, int k0) {
        float* A  = ldsA + buf * (BM * LDA);
        float* Bm = ldsB + buf * (BK * LDB);
#pragma unroll
        for (int j = 0; j < 4; ++j) {
            const int chunk = tid + j * 256;
            const int ar = chunk >> 3, ac = (chunk & 7) << 2;
            __builtin_amdgcn_global_load_async_to_lds_b128(
                (gv4i_t)(xb + (size_t)(tileM + ar) * I_SIZE + k0 + ac),
                (lv4i_t)(A + ar * LDA + ac), 0, 0);
            const int br = chunk >> 5, bc = (chunk & 31) << 2;
            __builtin_amdgcn_global_load_async_to_lds_b128(
                (gv4i_t)(wb + (size_t)(k0 + br) * O_SIZE + tileN + bc),
                (lv4i_t)(Bm + br * LDB + bc), 0, 0);
        }
    };
    auto sync_tiles = [&]() {
        wait_async_zero();
        __syncthreads();
    };
#else
    // ---- fallback staging through VGPRs ----
    float4 pa[4], pb[4];
    auto load_global = [&](int k0) {
#pragma unroll
        for (int j = 0; j < 4; ++j) {
            const int chunk = tid + j * 256;
            const int ar = chunk >> 3, ac = (chunk & 7) << 2;
            pa[j] = *(const float4*)(xb + (size_t)(tileM + ar) * I_SIZE + k0 + ac);
            const int br = chunk >> 5, bc = (chunk & 31) << 2;
            pb[j] = *(const float4*)(wb + (size_t)(k0 + br) * O_SIZE + tileN + bc);
        }
    };
    auto store_lds = [&](int buf) {
        float* A  = ldsA + buf * (BM * LDA);
        float* Bm = ldsB + buf * (BK * LDB);
#pragma unroll
        for (int j = 0; j < 4; ++j) {
            const int chunk = tid + j * 256;
            const int ar = chunk >> 3, ac = (chunk & 7) << 2;
            *(float4*)(A + ar * LDA + ac) = pa[j];
            const int br = chunk >> 5, bc = (chunk & 31) << 2;
            *(float4*)(Bm + br * LDB + bc) = pb[j];
        }
    };
#endif

    // ---- compute one K=32 step out of LDS: bf16 hi/lo split, 3 WMMAs each ----
    auto compute = [&](int buf) {
        const float* A  = ldsA + buf * (BM * LDA);
        const float* Bm = ldsB + buf * (BK * LDB);

        v16bf ahi[4], alo[4];
#pragma unroll
        for (int mi = 0; mi < 4; ++mi) {
            const int row = wy * 64 + mi * 16 + ln;
            const float* arow = A + row * LDA;
#pragma unroll
            for (int v = 0; v < 8; ++v) {
                const int kk = ((v & 4) << 2) + (hi16 << 3) + ((v & 3) << 1);
                const float2 p = *(const float2*)(arow + kk);
                const __bf16 h0 = (__bf16)p.x;
                const __bf16 h1 = (__bf16)p.y;
                ahi[mi][2 * v]     = h0;
                ahi[mi][2 * v + 1] = h1;
                alo[mi][2 * v]     = (__bf16)(p.x - (float)h0);
                alo[mi][2 * v + 1] = (__bf16)(p.y - (float)h1);
            }
        }

#pragma unroll
        for (int ni = 0; ni < 2; ++ni) {
            v16bf bhi, blo;
            const int col = wx * 32 + ni * 16 + ln;
#pragma unroll
            for (int v = 0; v < 8; ++v) {
                const int kk = ((v & 4) << 2) + (hi16 << 3) + ((v & 3) << 1);
                const float b0 = Bm[kk * LDB + col];
                const float b1 = Bm[(kk + 1) * LDB + col];
                const __bf16 h0 = (__bf16)b0;
                const __bf16 h1 = (__bf16)b1;
                bhi[2 * v]     = h0;
                bhi[2 * v + 1] = h1;
                blo[2 * v]     = (__bf16)(b0 - (float)h0);
                blo[2 * v + 1] = (__bf16)(b1 - (float)h1);
            }
#pragma unroll
            for (int mi = 0; mi < 4; ++mi) {
                acc[mi][ni] = __builtin_amdgcn_wmma_f32_16x16x32_bf16(
                    false, ahi[mi], false, bhi, (short)0, acc[mi][ni], false, false);
                acc[mi][ni] = __builtin_amdgcn_wmma_f32_16x16x32_bf16(
                    false, alo[mi], false, bhi, (short)0, acc[mi][ni], false, false);
                acc[mi][ni] = __builtin_amdgcn_wmma_f32_16x16x32_bf16(
                    false, ahi[mi], false, blo, (short)0, acc[mi][ni], false, false);
            }
        }
    };

    // ---- main K loop, double-buffered, one barrier per step ----
#if HAVE_TDM || HAVE_ASYNC_LDS
    issue_tiles(0, 0);
    sync_tiles();
#pragma unroll 1
    for (int kt = 0; kt < KTILES; ++kt) {
        const int cur = kt & 1;
        if (kt + 1 < KTILES) issue_tiles(cur ^ 1, (kt + 1) * BK);   // DMA overlaps compute
        compute(cur);
        if (kt + 1 < KTILES) sync_tiles();
    }
#else
    load_global(0);
    store_lds(0);
    __syncthreads();
#pragma unroll 1
    for (int kt = 0; kt < KTILES; ++kt) {
        const int cur = kt & 1;
        if (kt + 1 < KTILES) load_global((kt + 1) * BK);
        compute(cur);
        if (kt + 1 < KTILES) {
            store_lds(cur ^ 1);
            __syncthreads();
        }
    }
#endif

    // ---- epilogue: scatter accumulators ----
#pragma unroll
    for (int mi = 0; mi < 4; ++mi) {
        const int rowbase = tileM + wy * 64 + mi * 16 + hi16 * 8;
#pragma unroll
        for (int ni = 0; ni < 2; ++ni) {
            const int col = tileN + wx * 32 + ni * 16 + ln;
#pragma unroll
            for (int r = 0; r < 8; ++r)
                ob[(size_t)(rowbase + r) * O_SIZE + col] = acc[mi][ni][r];
        }
    }
}

extern "C" void kernel_launch(void* const* d_in, const int* in_sizes, int n_in,
                              void* d_out, int out_size, void* d_ws, size_t ws_size,
                              hipStream_t stream) {
    const float* x           = (const float*)d_in[0];
    const int*   domain_id   = (const int*)d_in[1];
    const float* fc_weight   = (const float*)d_in[2];
    const float* bias_weight = (const float*)d_in[3];
    float*       out         = (float*)d_out;

    const size_t lds_bytes = (size_t)(2 * BM * LDA + 2 * BK * LDB) * sizeof(float); // ~69 KB
    dim3 grid(O_SIZE / BN, T_SIZE / BM, B_SIZE);
    domain_linear_bf16x2_wmma<<<grid, 256, lds_bytes, stream>>>(
        x, domain_id, fc_weight, bias_weight, out);
}
